// BindingSiteNADPredictor_670014898390
// MI455X (gfx1250) — matrix operand order repaired and checked by reading, not compiled, not run on hardware
//
#include <hip/hip_runtime.h>

// ---------------------------------------------------------------------------
// BindingSiteNADPredictor for MI455X (gfx1250, wave32, WMMA).
// All dense GEMMs use v_wmma_f32_16x16x32_bf16 (f32 accumulate), 32 rows per
// block (2 row tiles x 8 waves) to amortize weight traffic and barriers.
// Weights are pre-packed into bf16 *fragment order* so the GEMM hot loop is:
// global_load_b128 -> ds_store_b128 -> ds_load_b128 -> v_wmma.
// Graph aggregation is done per-graph in LDS (edges are blocked per graph).
// ---------------------------------------------------------------------------

#define NUM_GRAPHS 8192
#define NODES_PER  24
#define DEG        8
#define NODE_DIM   1310
#define HID        256
#define ATT        128
#define HEADS      4
#define LAYERS     3
#define N_NODES    (NUM_GRAPHS * NODES_PER)        // 196608
#define N_EDGES    (N_NODES * DEG)                 // 1572864
#define EPG        (NODES_PER * DEG)               // 192 edges per graph
#define LN_EPS     1e-5f

// per-lane fragment record: 16 bf16 used + 8 pad => 48 B, 16B-aligned stride
#define FRAG_STRIDE 24

typedef __attribute__((ext_vector_type(16))) __bf16 v16bf;
typedef __attribute__((ext_vector_type(8)))  float  v8f;

union BF16Frag {
    unsigned short u[16];
    uint4          q[2];
    v16bf          v;
};

__device__ __forceinline__ unsigned short f2bf(float f) {
    // round-to-nearest-even fp32 -> bf16
    unsigned int u = __float_as_uint(f);
    u += 0x7FFFu + ((u >> 16) & 1u);
    return (unsigned short)(u >> 16);
}

__device__ __forceinline__ float waveReduceSum(float v) {
    #pragma unroll
    for (int off = 16; off > 0; off >>= 1) v += __shfl_xor(v, off, 32);
    return v;
}

// ---------------------------------------------------------------------------
// Weight pre-pack: f32 row-major [K x NC]  ->  bf16 fragment order
//   out index = ((kblock*NT + nt)*32 + lane)*16 + u
//   where kp = (lane>>4)*8 + (u&7) + (u>>3)*16, k = kblock*32+kp, n = nt*16 + (lane&15)
// (B-fragment layout matching A: lanes 0-15 hold K={0..7,16..23},
//  lanes 16-31 hold K={8..15,24..31}.)
// ---------------------------------------------------------------------------
__global__ __launch_bounds__(256)
void pack_weights_kernel(const float* __restrict__ W, unsigned short* __restrict__ Wf,
                         int K, int NC)
{
    int idx = blockIdx.x * 256 + threadIdx.x;
    int total = ((K + 31) / 32) * NC * 32;
    if (idx >= total) return;
    int u    = idx & 15;
    int lane = (idx >> 4) & 31;
    int rec  = idx >> 9;              // kblock*NT + nt
    int NT   = NC / 16;
    int nt   = rec % NT;
    int kb   = rec / NT;
    int kp   = ((lane >> 4) << 3) + (u & 7) + ((u >> 3) << 4);
    int k    = kb * 32 + kp;
    int n    = nt * 16 + (lane & 15);
    float f  = (k < K) ? W[(size_t)k * NC + n] : 0.0f;
    Wf[idx] = f2bf(f);
}

// ---------------------------------------------------------------------------
// Core: (16*RTILES)-row x NCOLS GEMM accumulate. blockDim.x == 256 (8 waves).
// A: row-major [nrows x K] f32 (converted to bf16 inline).
// Wf: pre-packed bf16 fragments. acc: [RTILES][NCOLS/128] per wave, flat.
// ---------------------------------------------------------------------------
template<int NCOLS, int RTILES>
__device__ __forceinline__ void gemm_acc(const float* __restrict__ A,
                                         const unsigned short* __restrict__ Wf,
                                         int K, int row0,
                                         v8f* acc,
                                         unsigned short* __restrict__ AsF,
                                         unsigned short* __restrict__ BsF)
{
    const int tid  = threadIdx.x;
    const int lane = tid & 31;
    const int wave = tid >> 5;
    constexpr int TPW = NCOLS / 128;   // column tiles per wave
    constexpr int NT  = NCOLS / 16;    // column tiles total

    // A staging geometry (fixed per thread): one bf16 pair per row tile/step.
    // A-fragment layout: lanes 0-15: M=lane, K in {0..7,16..23};
    //                    lanes 16-31: M=lane-16, K in {8..15,24..31}.
    const int am    = tid >> 4;
    const int akp   = (tid & 15) * 2;
    const int alane = am + 16 * ((akp >> 3) & 1);
    const int au    = (akp & 7) | ((akp >> 4) << 3);
    const float*  aSrc[RTILES];
    unsigned int* aDst[RTILES];
    #pragma unroll
    for (int rt = 0; rt < RTILES; rt++) {
        aSrc[rt] = A + (size_t)(row0 + rt * 16 + am) * K + akp;
        aDst[rt] = (unsigned int*)(AsF + (rt * 32 + alane) * FRAG_STRIDE + au);
    }

    const int nsteps = (K + 31) >> 5;
    for (int s = 0; s < nsteps; s++) {
        const int k0   = s * 32;
        const int kmax = K - k0;
        __syncthreads();
        // --- stage A: float2 load, pack to one b32 LDS store, per row tile
        #pragma unroll
        for (int rt = 0; rt < RTILES; rt++) {
            float2 f = make_float2(0.0f, 0.0f);
            if (kmax >= 32 || akp + 1 < kmax) {
                f = *(const float2*)(aSrc[rt] + k0);
            } else if (akp < kmax) {
                f.x = aSrc[rt][k0];
            }
            *aDst[rt] = (unsigned int)f2bf(f.x) | ((unsigned int)f2bf(f.y) << 16);
        }
        // --- stage B: straight uint4 copy of pre-packed fragments
        {
            const uint4* src = (const uint4*)(Wf + (size_t)s * NCOLS * 32);
            for (int i = tid; i < NT * 64; i += 256) {
                int rec = i >> 1, half = i & 1;
                uint4 v = src[i];
                *(uint4*)(BsF + rec * FRAG_STRIDE + half * 8) = v;
            }
        }
        __syncthreads();
        // --- fragments (2x ds_load_b128 each) + WMMA
        BF16Frag fa[RTILES];
        #pragma unroll
        for (int rt = 0; rt < RTILES; rt++) {
            const uint4* ap = (const uint4*)(AsF + (rt * 32 + lane) * FRAG_STRIDE);
            fa[rt].q[0] = ap[0];
            fa[rt].q[1] = ap[1];
        }
        #pragma unroll
        for (int t = 0; t < TPW; t++) {
            const int nt = wave * TPW + t;
            BF16Frag fb;
            const uint4* bp = (const uint4*)(BsF + (nt * 32 + lane) * FRAG_STRIDE);
            fb.q[0] = bp[0];
            fb.q[1] = bp[1];
            #pragma unroll
            for (int rt = 0; rt < RTILES; rt++) {
                acc[rt * TPW + t] = __builtin_amdgcn_wmma_f32_16x16x32_bf16(
                    false, fa[rt].v, false, fb.v, (short)0, acc[rt * TPW + t], false, false);
            }
        }
    }
}

// Spill accumulators to LDS (f32 (16*RTILES) x NCOLS) for row-wise epilogues.
// C layout: VGPR v -> M = v + 8*(lane>=16), N = lane&15.
template<int NCOLS, int RTILES>
__device__ __forceinline__ void spill_acc(const v8f* acc, float* Cs)
{
    const int lane = threadIdx.x & 31;
    const int wave = threadIdx.x >> 5;
    constexpr int TPW = NCOLS / 128;
    const int n  = lane & 15;
    const int mb = (lane >> 4) * 8;
    #pragma unroll
    for (int rt = 0; rt < RTILES; rt++) {
        #pragma unroll
        for (int t = 0; t < TPW; t++) {
            int n0 = (wave * TPW + t) * 16;
            #pragma unroll
            for (int v = 0; v < 8; v++) {
                Cs[(rt * 16 + mb + v) * NCOLS + n0 + n] = acc[rt * TPW + t][v];
            }
        }
    }
    __syncthreads();
}

// ---------------------------------------------------------------------------
// Kernel 1: h0 = relu(layernorm(x @ Wp + bp, g0, beta0))      grid = N/32
// ---------------------------------------------------------------------------
__global__ __launch_bounds__(256)
void proj_ln_kernel(const float* __restrict__ x, const unsigned short* __restrict__ WpF,
                    const float* __restrict__ bp, const float* __restrict__ g0,
                    const float* __restrict__ beta0, float* __restrict__ h)
{
    __shared__ __align__(16) unsigned short AsF[2 * 32 * FRAG_STRIDE];
    __shared__ __align__(16) unsigned short BsF[(HID / 16) * 32 * FRAG_STRIDE];
    __shared__ float Cs[32 * HID];

    v8f acc[2 * (HID / 128)] = {};
    const int row0 = blockIdx.x * 32;
    gemm_acc<HID, 2>(x, WpF, NODE_DIM, row0, acc, AsF, BsF);
    spill_acc<HID, 2>(acc, Cs);

    const int lane = threadIdx.x & 31, wave = threadIdx.x >> 5;
    #pragma unroll
    for (int rr = 0; rr < 4; rr++) {
        int r = 4 * wave + rr;
        float vals[8], s = 0.f, s2 = 0.f;
        #pragma unroll
        for (int j = 0; j < 8; j++) {
            int c = lane + 32 * j;
            float v = Cs[r * HID + c] + bp[c];
            vals[j] = v; s += v; s2 += v * v;
        }
        s  = waveReduceSum(s);
        s2 = waveReduceSum(s2);
        float mu  = s * (1.0f / HID);
        float var = s2 * (1.0f / HID) - mu * mu;
        float rs  = rsqrtf(var + LN_EPS);
        #pragma unroll
        for (int j = 0; j < 8; j++) {
            int c = lane + 32 * j;
            float v = (vals[j] - mu) * rs * g0[c] + beta0[c];
            h[(size_t)(row0 + r) * HID + c] = fmaxf(v, 0.0f);
        }
    }
}

// ---------------------------------------------------------------------------
// Generic GEMM: Out = [relu](A @ W [+ bias]); direct register->global store.
// grid = nrows/32.
// ---------------------------------------------------------------------------
template<int NCOLS, bool RELU, bool HASBIAS>
__global__ __launch_bounds__(256)
void gemm_kernel(const float* __restrict__ A, const unsigned short* __restrict__ Wf,
                 const float* __restrict__ bias, float* __restrict__ Out, int K)
{
    __shared__ __align__(16) unsigned short AsF[2 * 32 * FRAG_STRIDE];
    __shared__ __align__(16) unsigned short BsF[(NCOLS / 16) * 32 * FRAG_STRIDE];

    constexpr int TPW = NCOLS / 128;
    v8f acc[2 * TPW] = {};
    const int row0 = blockIdx.x * 32;
    gemm_acc<NCOLS, 2>(A, Wf, K, row0, acc, AsF, BsF);

    const int lane = threadIdx.x & 31, wave = threadIdx.x >> 5;
    const int n  = lane & 15;
    const int mb = (lane >> 4) * 8;
    #pragma unroll
    for (int rt = 0; rt < 2; rt++) {
        #pragma unroll
        for (int t = 0; t < TPW; t++) {
            int col = (wave * TPW + t) * 16 + n;
            float b = HASBIAS ? bias[col] : 0.0f;
            #pragma unroll
            for (int v = 0; v < 8; v++) {
                float val = acc[rt * TPW + t][v] + b;
                if (RELU) val = fmaxf(val, 0.0f);
                Out[(size_t)(row0 + rt * 16 + mb + v) * NCOLS + col] = val;
            }
        }
    }
}

// ---------------------------------------------------------------------------
// Per-graph GCN aggregation + bias + residual + LN + relu.   grid = G
// ---------------------------------------------------------------------------
__global__ __launch_bounds__(256)
void agg_ln_kernel(const float* __restrict__ hW, const float* __restrict__ hPrev,
                   const int* __restrict__ src, const int* __restrict__ dst,
                   const float* __restrict__ ew,
                   const float* __restrict__ bg, const float* __restrict__ lng,
                   const float* __restrict__ lnb, float* __restrict__ hOut,
                   int useResidual)
{
    __shared__ float hw[NODES_PER * HID];    // 24 KB
    __shared__ float agg[NODES_PER * HID];   // 24 KB
    __shared__ float degL[NODES_PER];
    __shared__ int   eS[EPG];
    __shared__ int   eD[EPG];
    __shared__ float eN[EPG];

    const int g = blockIdx.x;
    const int tid = threadIdx.x, lane = tid & 31, wave = tid >> 5;
    const size_t nodeBase = (size_t)g * NODES_PER;

    for (int i = tid; i < NODES_PER * HID; i += 256) {
        hw[i]  = hW[nodeBase * HID + i];
        agg[i] = 0.0f;
    }
    if (tid < NODES_PER) degL[tid] = 1.0f;   // deg = sum(ew at dst) + 1
    __syncthreads();

    if (tid < EPG) {
        int e = g * EPG + tid;
        int s = src[e] - (int)nodeBase;
        int d = dst[e] - (int)nodeBase;
        float w = ew[e];
        eS[tid] = s; eD[tid] = d; eN[tid] = w;
        atomicAdd(&degL[d], w);              // ds_add_f32
    }
    __syncthreads();
    if (tid < EPG) {
        eN[tid] = rsqrtf(degL[eS[tid]]) * eN[tid] * rsqrtf(degL[eD[tid]]);
    }
    __syncthreads();

    // Each wave owns a disjoint 32-channel slice -> serial edge walk is race-free.
    {
        const int c = wave * 32 + lane;
        for (int e = 0; e < EPG; e++) {
            agg[eD[e] * HID + c] += hw[eS[e] * HID + c] * eN[e];
        }
    }
    __syncthreads();

    // Epilogue: + hW/deg + bias (+ residual), layernorm, relu. Wave per row.
    for (int r = wave; r < NODES_PER; r += 8) {
        float dinv = 1.0f / degL[r];
        float vals[8], s = 0.f, s2 = 0.f;
        #pragma unroll
        for (int j = 0; j < 8; j++) {
            int c = lane + 32 * j;
            float v = agg[r * HID + c] + hw[r * HID + c] * dinv + bg[c];
            if (useResidual) v += hPrev[(nodeBase + r) * HID + c];
            vals[j] = v; s += v; s2 += v * v;
        }
        s  = waveReduceSum(s);
        s2 = waveReduceSum(s2);
        float mu  = s * (1.0f / HID);
        float var = s2 * (1.0f / HID) - mu * mu;
        float rs  = rsqrtf(var + LN_EPS);
        #pragma unroll
        for (int j = 0; j < 8; j++) {
            int c = lane + 32 * j;
            float v = (vals[j] - mu) * rs * lng[c] + lnb[c];
            hOut[(nodeBase + r) * HID + c] = fmaxf(v, 0.0f);
        }
    }
}

// ---------------------------------------------------------------------------
// scores = tanh(M @ Wa1 + ba1) @ Wa2 + ba2                    grid = N/32
// ---------------------------------------------------------------------------
__global__ __launch_bounds__(256)
void att_score_kernel(const float* __restrict__ M, const unsigned short* __restrict__ Wa1F,
                      const float* __restrict__ ba1, const float* __restrict__ Wa2,
                      const float* __restrict__ ba2, float* __restrict__ scores)
{
    __shared__ __align__(16) unsigned short AsF[2 * 32 * FRAG_STRIDE];
    __shared__ __align__(16) unsigned short BsF[(ATT / 16) * 32 * FRAG_STRIDE];
    __shared__ float Cs[32 * ATT];

    v8f acc[2 * (ATT / 128)] = {};
    const int row0 = blockIdx.x * 32;
    gemm_acc<ATT, 2>(M, Wa1F, HID, row0, acc, AsF, BsF);
    spill_acc<ATT, 2>(acc, Cs);

    const int tid = threadIdx.x, lane = tid & 31, wave = tid >> 5;
    for (int i = tid; i < 32 * ATT; i += 256) {
        Cs[i] = tanhf(Cs[i] + ba1[i & (ATT - 1)]);
    }
    __syncthreads();

    #pragma unroll
    for (int rr = 0; rr < 4; rr++) {
        int r = 4 * wave + rr;
        float a[HEADS] = {0.f, 0.f, 0.f, 0.f};
        #pragma unroll
        for (int j = 0; j < ATT / 32; j++) {
            int c = lane + 32 * j;
            float t = Cs[r * ATT + c];
            #pragma unroll
            for (int hd = 0; hd < HEADS; hd++) a[hd] += t * Wa2[c * HEADS + hd];
        }
        #pragma unroll
        for (int hd = 0; hd < HEADS; hd++) a[hd] = waveReduceSum(a[hd]);
        if (lane == 0) {
            size_t o = (size_t)(row0 + r) * HEADS;
            #pragma unroll
            for (int hd = 0; hd < HEADS; hd++) scores[o + hd] = a[hd] + ba2[hd];
        }
    }
}

// ---------------------------------------------------------------------------
// Per-graph segment softmax + attention/mean pooling -> emb (G x 512)
// ---------------------------------------------------------------------------
__global__ __launch_bounds__(256)
void pool_kernel(const float* __restrict__ M, const float* __restrict__ scores,
                 float* __restrict__ emb)
{
    __shared__ float sc[NODES_PER * HEADS];
    __shared__ float wmean[NODES_PER];
    const int g = blockIdx.x, tid = threadIdx.x;
    const size_t base = (size_t)g * NODES_PER;

    if (tid < NODES_PER * HEADS) sc[tid] = scores[base * HEADS + tid];
    __syncthreads();
    if (tid < HEADS) {
        float mx = -1e30f;
        for (int n = 0; n < NODES_PER; n++) mx = fmaxf(mx, sc[n * HEADS + tid]);
        float sum = 0.f;
        for (int n = 0; n < NODES_PER; n++) {
            float e = expf(sc[n * HEADS + tid] - mx);
            sc[n * HEADS + tid] = e; sum += e;
        }
        float inv = 1.0f / sum;
        for (int n = 0; n < NODES_PER; n++) sc[n * HEADS + tid] *= inv;
    }
    __syncthreads();
    if (tid < NODES_PER) {
        float m = 0.f;
        #pragma unroll
        for (int hd = 0; hd < HEADS; hd++) m += sc[tid * HEADS + hd];
        wmean[tid] = m * (1.0f / HEADS);
    }
    __syncthreads();

    float a = 0.f, p = 0.f;
    for (int n = 0; n < NODES_PER; n++) {
        float m = M[(base + n) * HID + tid];
        a += m * wmean[n];
        p += m;
    }
    emb[(size_t)g * (2 * HID) + tid]       = a;                      // att_pooled
    emb[(size_t)g * (2 * HID) + HID + tid] = p * (1.0f / NODES_PER); // mean pool
}

// ---------------------------------------------------------------------------
// out = z2 @ Wc3 + bc3  (128 -> 2).  Wave per row, grid = G/8.
// ---------------------------------------------------------------------------
__global__ __launch_bounds__(256)
void head_kernel(const float* __restrict__ z2, const float* __restrict__ Wc3,
                 const float* __restrict__ bc3, float* __restrict__ out)
{
    const int lane = threadIdx.x & 31;
    const int row  = blockIdx.x * 8 + (threadIdx.x >> 5);
    float a0 = 0.f, a1 = 0.f;
    #pragma unroll
    for (int j = 0; j < 4; j++) {
        int c = lane + 32 * j;
        float v = z2[(size_t)row * 128 + c];
        a0 += v * Wc3[c * 2 + 0];
        a1 += v * Wc3[c * 2 + 1];
    }
    a0 = waveReduceSum(a0);
    a1 = waveReduceSum(a1);
    if (lane == 0) {
        out[row * 2 + 0] = a0 + bc3[0];
        out[row * 2 + 1] = a1 + bc3[1];
    }
}

// ---------------------------------------------------------------------------
extern "C" void kernel_launch(void* const* d_in, const int* in_sizes, int n_in,
                              void* d_out, int out_size, void* d_ws, size_t ws_size,
                              hipStream_t stream)
{
    const float* x          = (const float*)d_in[0];
    const int*   edge_index = (const int*)  d_in[1];
    const float* edge_attr  = (const float*)d_in[2];
    // d_in[3] = batch (implicit in per-graph blocking, unused)
    const float* Wp    = (const float*)d_in[4];
    const float* bp    = (const float*)d_in[5];
    const float* g0    = (const float*)d_in[6];
    const float* beta0 = (const float*)d_in[7];
    const float* Wg    = (const float*)d_in[8];   // (3,256,256)
    const float* bg    = (const float*)d_in[9];   // (3,256)
    const float* lng   = (const float*)d_in[10];
    const float* lnb   = (const float*)d_in[11];
    const float* Wa1   = (const float*)d_in[12];
    const float* ba1   = (const float*)d_in[13];
    const float* Wa2   = (const float*)d_in[14];
    const float* ba2   = (const float*)d_in[15];
    const float* Wc1   = (const float*)d_in[16];
    const float* bc1   = (const float*)d_in[17];
    const float* Wc2   = (const float*)d_in[18];
    const float* bc2   = (const float*)d_in[19];
    const float* Wc3   = (const float*)d_in[20];
    const float* bc3   = (const float*)d_in[21];

    const int* src = edge_index;
    const int* dst = edge_index + N_EDGES;

    // ---- workspace layout ----
    float* ws     = (float*)d_ws;
    float* buf0   = ws;                                    // N x 256
    float* buf1   = buf0   + (size_t)N_NODES * HID;        // N x 256
    float* bufW   = buf1   + (size_t)N_NODES * HID;        // N x 256 (h @ Wg)
    float* scores = bufW   + (size_t)N_NODES * HID;        // N x 4
    float* emb    = scores + (size_t)N_NODES * HEADS;      // G x 512
    float* z1     = emb    + (size_t)NUM_GRAPHS * 2 * HID; // G x 256
    float* z2     = z1     + (size_t)NUM_GRAPHS * HID;     // G x 128
    // bf16 fragment-packed weight arena (16B-aligned: all offsets are /16)
    unsigned short* uArena = (unsigned short*)(z2 + (size_t)NUM_GRAPHS * ATT);
    const size_t szWp  = (size_t)((NODE_DIM + 31) / 32) * HID * 32;  // 41*8192
    const size_t szWg  = (size_t)(HID / 32) * HID * 32;              // 65536
    const size_t szWa1 = (size_t)(HID / 32) * ATT * 32;              // 32768
    const size_t szWc1 = (size_t)(2 * HID / 32) * HID * 32;          // 131072
    const size_t szWc2 = (size_t)(HID / 32) * ATT * 32;              // 32768
    unsigned short* WpF  = uArena;
    unsigned short* WgF  = WpF  + szWp;                    // 3 consecutive
    unsigned short* Wa1F = WgF  + 3 * szWg;
    unsigned short* Wc1F = Wa1F + szWa1;
    unsigned short* Wc2F = Wc1F + szWc1;

    const dim3 blk(256);

    // 0) pack all GEMM weights to bf16 fragment order (cheap, deterministic)
    {
        auto pack = [&](const float* W, unsigned short* Wf, int K, int NC) {
            int total = ((K + 31) / 32) * NC * 32;
            pack_weights_kernel<<<(total + 255) / 256, blk, 0, stream>>>(W, Wf, K, NC);
        };
        pack(Wp, WpF, NODE_DIM, HID);
        for (int l = 0; l < LAYERS; l++)
            pack(Wg + (size_t)l * HID * HID, WgF + (size_t)l * szWg, HID, HID);
        pack(Wa1, Wa1F, HID, ATT);
        pack(Wc1, Wc1F, 2 * HID, HID);
        pack(Wc2, Wc2F, HID, ATT);
    }

    // 1) projection + LN + relu
    proj_ln_kernel<<<N_NODES / 32, blk, 0, stream>>>(x, WpF, bp, g0, beta0, buf0);

    // 2) GCN layers (ping-pong buf0/buf1)
    const float* hin = buf0;
    float* hout = buf1;
    for (int l = 0; l < LAYERS; l++) {
        gemm_kernel<HID, false, false><<<N_NODES / 32, blk, 0, stream>>>(
            hin, WgF + (size_t)l * szWg, nullptr, bufW, HID);
        agg_ln_kernel<<<NUM_GRAPHS, blk, 0, stream>>>(
            bufW, hin, src, dst, edge_attr,
            bg + (size_t)l * HID, lng + (size_t)l * HID, lnb + (size_t)l * HID,
            hout, (l > 0) ? 1 : 0);
        float* t = (float*)hin; hin = hout; hout = t;
    }
    const float* M = hin;   // final node embeddings

    // 3) attention scores + per-graph softmax pooling
    att_score_kernel<<<N_NODES / 32, blk, 0, stream>>>(M, Wa1F, ba1, Wa2, ba2, scores);
    pool_kernel<<<NUM_GRAPHS, blk, 0, stream>>>(M, scores, emb);

    // 4) classifier
    gemm_kernel<HID, true, true><<<NUM_GRAPHS / 32, blk, 0, stream>>>(emb, Wc1F, bc1, z1, 2 * HID);
    gemm_kernel<ATT, true, true><<<NUM_GRAPHS / 32, blk, 0, stream>>>(z1, Wc2F, bc2, z2, HID);
    head_kernel<<<NUM_GRAPHS / 8, blk, 0, stream>>>(z2, Wc3, bc3, (float*)d_out);
}
